// SpikingSelfAttention_27530740367503
// MI455X (gfx1250) — compile-verified, matrix operand order checked by reading, not run on hardware
//
#include <hip/hip_runtime.h>
#include <hip/hip_bf16.h>

// ---------------------------------------------------------------------------
// Spiking self-attention for MI455X (gfx1250, wave32, WMMA 16x16x32 f16).
// T=4, B=32, C=512, N=256, heads=8, d=64.
// All GEMMs use v_wmma_f32_16x16x32_f16 with 2x2 register blocking:
// one wave owns a 32x32 output tile, so each A/B fragment pair feeds four
// WMMAs (2 b128 loads per WMMA).  LIF membrane state lives in registers
// across the T loop inside the fused GEMM kernels, never touching memory.
// __launch_bounds__(256,1) lets the allocator hold accumulators + LIF state
// without scratch spills (bandwidth-bound: occupancy is cheap, spills aren't).
// ---------------------------------------------------------------------------

typedef __attribute__((ext_vector_type(16))) _Float16 v16h;
typedef __attribute__((ext_vector_type(8)))  float    v8f;

#define TT 4
#define BB 32
#define CC 512
#define NN 256
#define HH 8
#define DD 64

// Fragment loader: for A (16x32, row-major, lda elems) lane<16 holds row=lane,
// K=k0..k0+15; lane>=16 holds row=lane-16, K=k0+16..k0+31. The B operand uses
// the mirrored layout, i.e. load from B^T (columns contiguous along K).
__device__ __forceinline__ v16h load_frag(const _Float16* base, int lda, int k0, int lane) {
    int row = lane & 15;
    int kh  = (lane >> 4) << 4;
    return *reinterpret_cast<const v16h*>(base + (size_t)row * lda + k0 + kh);
}

__device__ __forceinline__ v8f wmma_f16(v16h a, v16h b, v8f c) {
    return __builtin_amdgcn_wmma_f32_16x16x32_f16(false, a, false, b, (short)0, c, false, false);
}

// ---------------------------------------------------------------------------
// Kernel 0: weight f32->f16 + BN folding.
// bnscale/bnshift: [3][512] for q,k,v.  pscale/pshift: [512] (p_bias folded).
// ---------------------------------------------------------------------------
__global__ void prep_kernel(const float* qw, const float* qg, const float* qb, const float* qm, const float* qv,
                            const float* kw, const float* kg, const float* kb, const float* km, const float* kv,
                            const float* vw, const float* vg, const float* vb, const float* vm, const float* vv,
                            const float* pw, const float* pbias, const float* pg, const float* pb,
                            const float* pm, const float* pv,
                            _Float16* wq, _Float16* wk, _Float16* wv_, _Float16* wp,
                            float* bnscale, float* bnshift, float* pscale, float* pshift) {
    int id = blockIdx.x * 256 + threadIdx.x;            // 262144 threads
    wq[id]  = (_Float16)qw[id];
    wk[id]  = (_Float16)kw[id];
    wv_[id] = (_Float16)vw[id];
    wp[id]  = (_Float16)pw[id];
    if (id < CC) {
        float s;
        s = qg[id] * rsqrtf(qv[id] + 1e-5f); bnscale[id]        = s; bnshift[id]        = qb[id] - qm[id] * s;
        s = kg[id] * rsqrtf(kv[id] + 1e-5f); bnscale[CC + id]   = s; bnshift[CC + id]   = kb[id] - km[id] * s;
        s = vg[id] * rsqrtf(vv[id] + 1e-5f); bnscale[2*CC + id] = s; bnshift[2*CC + id] = vb[id] - vm[id] * s;
        s = pg[id] * rsqrtf(pv[id] + 1e-5f); pscale[id] = s;
        pshift[id] = (pbias[id] - pm[id]) * s + pb[id];
    }
}

// ---------------------------------------------------------------------------
// Kernel 1: input LIF (vth=1, tau=2, hard reset) over T; write xsT [T,B,N,C] f16
// (B-operand layout for the QKV GEMM: K=C contiguous).
// ---------------------------------------------------------------------------
__global__ void lif_input_kernel(const float* x, _Float16* xsT) {
    size_t id = (size_t)blockIdx.x * 256 + threadIdx.x; // B*C*N = 4194304 threads
    int n = (int)(id & (NN - 1));
    size_t r = id >> 8;
    int c = (int)(r & (CC - 1));
    int b = (int)(r >> 9);
    float v = 0.0f;
#pragma unroll
    for (int t = 0; t < TT; ++t) {
        float xi = x[(((size_t)t * BB + b) * CC + c) * NN + n];
        v = 0.5f * (v + xi);                 // v += (x - v)/tau, tau=2
        float s = (v >= 1.0f) ? 1.0f : 0.0f;
        v *= (1.0f - s);                     // hard reset
        xsT[(((size_t)t * BB + b) * NN + n) * CC + c] = (_Float16)s;
    }
}

// ---------------------------------------------------------------------------
// Kernel 2: fused QKV GEMM + BN + LIF.  One wave = one 32x32 (c_out x n) tile
// (2x2 WMMA blocking), loops t=0..3 with LIF state in registers.
// 64 WMMAs per t (K=512).  BN constants are re-read in the epilogue (L2-hot)
// instead of being held live across the K-loop.
//   q spikes -> [T,B,h,N,d]   (A-operand layout for attention GEMM)
//   k,v spikes -> [T,B,h,d,N] (operand layouts for kv^T GEMM)
// waves = 3 * 32 * 16 * 8 = 12288 -> 1536 blocks of 256.
// ---------------------------------------------------------------------------
__global__ void __launch_bounds__(256, 1)
qkv_gemm_lif_kernel(const _Float16* xsT,
                    const _Float16* wq, const _Float16* wk, const _Float16* wv,
                    const float* bnscale, const float* bnshift,
                    _Float16* qs, _Float16* ks, _Float16* vs) {
    int w = blockIdx.x * 8 + (threadIdx.x >> 5);
    int lane = threadIdx.x & 31;
    int ntile  = w & 7;         w >>= 3;   // 8 tiles of 32 cols (N)
    int ctile  = w & 15;        w >>= 4;   // 16 tiles of 32 rows (C)
    int b      = w & 31;        w >>= 5;
    int branch = w;                        // 0=q 1=k 2=v
    const _Float16* W = (branch == 0) ? wq : (branch == 1) ? wk : wv;
    int c0 = ctile * 32, n0 = ntile * 32;
    int colL = lane & 15;
    int rowOff = (lane >> 4) * 8;

    float vmem[2][2][8];
#pragma unroll
    for (int mi = 0; mi < 2; ++mi)
#pragma unroll
        for (int ni = 0; ni < 2; ++ni)
#pragma unroll
            for (int r = 0; r < 8; ++r) vmem[mi][ni][r] = 0.0f;

    const _Float16* A0 = W + (size_t)c0 * CC;
    for (int t = 0; t < TT; ++t) {
        const _Float16* B0 = xsT + ((((size_t)t * BB + b) * NN + n0) * CC);
        v8f acc[2][2];
#pragma unroll
        for (int mi = 0; mi < 2; ++mi)
#pragma unroll
            for (int ni = 0; ni < 2; ++ni) acc[mi][ni] = (v8f){};
#pragma unroll
        for (int k0 = 0; k0 < CC; k0 += 32) {
            v16h a0 = load_frag(A0,            CC, k0, lane);
            v16h a1 = load_frag(A0 + 16 * CC,  CC, k0, lane);
            v16h b0 = load_frag(B0,            CC, k0, lane);
            v16h b1 = load_frag(B0 + 16 * CC,  CC, k0, lane);
            acc[0][0] = wmma_f16(a0, b0, acc[0][0]);
            acc[0][1] = wmma_f16(a0, b1, acc[0][1]);
            acc[1][0] = wmma_f16(a1, b0, acc[1][0]);
            acc[1][1] = wmma_f16(a1, b1, acc[1][1]);
        }
        size_t tb = (size_t)t * BB + b;
#pragma unroll
        for (int mi = 0; mi < 2; ++mi)
#pragma unroll
            for (int ni = 0; ni < 2; ++ni)
#pragma unroll
                for (int r = 0; r < 8; ++r) {
                    int c = c0 + mi * 16 + r + rowOff;
                    int n = n0 + ni * 16 + colL;
                    float y = acc[mi][ni][r] * bnscale[branch * CC + c]
                            + bnshift[branch * CC + c];             // BN (eval)
                    float v = 0.5f * (vmem[mi][ni][r] + y);         // LIF, tau=2
                    float s = (v >= 1.0f) ? 1.0f : 0.0f;            // vth = 1
                    vmem[mi][ni][r] = v * (1.0f - s);
                    _Float16 sp = (_Float16)s;
                    if (branch == 0) {
                        int h = c >> 6, dd = c & 63;
                        qs[(((tb * HH + h) * NN + n) << 6) + dd] = sp;  // [T,B,h,N,d]
                    } else {
                        _Float16* dst = (branch == 1) ? ks : vs;
                        dst[((tb * CC + c) << 8) + n] = sp;             // [T,B,h,d,N]
                    }
                }
    }
}

// ---------------------------------------------------------------------------
// Kernel 3: kvT[e,d] = sum_n v[n,e] * k[n,d] per (t,b,h).  M=e, cols=d, K=n=256.
// One wave = 32x32 tile (2x2 blocking), 32 WMMAs.  D stored row-major [64,64]
// f16 = B-operand layout for the attention GEMM.
// waves = 4*32*8*4 = 4096 -> 512 blocks.
// ---------------------------------------------------------------------------
__global__ void __launch_bounds__(256, 1)
kv_gemm_kernel(const _Float16* ks, const _Float16* vs, _Float16* kvT) {
    int w = blockIdx.x * 8 + (threadIdx.x >> 5);
    int lane = threadIdx.x & 31;
    int tile = w & 3;   w >>= 2;   // 2x2 tiles of 32x32 over [64,64]
    int h    = w & 7;   w >>= 3;
    int b    = w & 31;  w >>= 5;
    int t    = w;
    int e0 = (tile >> 1) * 32;
    int d0 = (tile & 1) * 32;
    size_t base = (((size_t)t * BB + b) * CC + h * DD);
    const _Float16* A0 = vs + (base + e0) * NN;   // rows e, K=n
    const _Float16* B0 = ks + (base + d0) * NN;   // rows d, K=n
    v8f acc[2][2];
#pragma unroll
    for (int mi = 0; mi < 2; ++mi)
#pragma unroll
        for (int ni = 0; ni < 2; ++ni) acc[mi][ni] = (v8f){};
#pragma unroll
    for (int k0 = 0; k0 < NN; k0 += 32) {
        v16h a0 = load_frag(A0,            NN, k0, lane);
        v16h a1 = load_frag(A0 + 16 * NN,  NN, k0, lane);
        v16h b0 = load_frag(B0,            NN, k0, lane);
        v16h b1 = load_frag(B0 + 16 * NN,  NN, k0, lane);
        acc[0][0] = wmma_f16(a0, b0, acc[0][0]);
        acc[0][1] = wmma_f16(a0, b1, acc[0][1]);
        acc[1][0] = wmma_f16(a1, b0, acc[1][0]);
        acc[1][1] = wmma_f16(a1, b1, acc[1][1]);
    }
    _Float16* D = kvT + ((((size_t)t * BB + b) * HH + h) << 12);
    int colL = lane & 15, rowOff = (lane >> 4) * 8;
#pragma unroll
    for (int mi = 0; mi < 2; ++mi)
#pragma unroll
        for (int ni = 0; ni < 2; ++ni)
#pragma unroll
            for (int r = 0; r < 8; ++r)
                D[(e0 + mi * 16 + r + rowOff) * DD + d0 + ni * 16 + colL] =
                    (_Float16)acc[mi][ni][r];
}

// ---------------------------------------------------------------------------
// Kernel 4: a[n,e] = 0.125 * sum_d q[n,d]*kvT[e,d]; attention LIF (vth=0.5)
// with v-state in registers across t.  One wave = 32x32 (n x e) tile,
// 8 WMMAs per t (K=64).  Spikes -> ss [T,B,N,C] f16 (B-operand layout for the
// output projection).  waves = 32*8*8*2 = 4096 -> 512 blocks.
// ---------------------------------------------------------------------------
__global__ void __launch_bounds__(256, 1)
attn_lif_kernel(const _Float16* qs, const _Float16* kvT, _Float16* ss) {
    int w = blockIdx.x * 8 + (threadIdx.x >> 5);
    int lane = threadIdx.x & 31;
    int et = w & 1;   w >>= 1;   // 2 tiles of 32 cols (e)
    int nt = w & 7;   w >>= 3;   // 8 tiles of 32 rows (n)
    int h  = w & 7;   w >>= 3;
    int b  = w;
    int n0 = nt * 32, e0 = et * 32;
    int colL = lane & 15, rowOff = (lane >> 4) * 8;
    float vmem[2][2][8];
#pragma unroll
    for (int mi = 0; mi < 2; ++mi)
#pragma unroll
        for (int ni = 0; ni < 2; ++ni)
#pragma unroll
            for (int r = 0; r < 8; ++r) vmem[mi][ni][r] = 0.0f;

    for (int t = 0; t < TT; ++t) {
        size_t tbh = ((size_t)t * BB + b) * HH + h;
        const _Float16* A0 = qs  + (tbh * NN + n0) * DD;   // rows n, K=d
        const _Float16* B0 = kvT + (tbh << 12) + e0 * DD;  // rows e, K=d
        v8f acc[2][2];
#pragma unroll
        for (int mi = 0; mi < 2; ++mi)
#pragma unroll
            for (int ni = 0; ni < 2; ++ni) acc[mi][ni] = (v8f){};
#pragma unroll
        for (int k0 = 0; k0 < DD; k0 += 32) {
            v16h a0 = load_frag(A0,            DD, k0, lane);
            v16h a1 = load_frag(A0 + 16 * DD,  DD, k0, lane);
            v16h b0 = load_frag(B0,            DD, k0, lane);
            v16h b1 = load_frag(B0 + 16 * DD,  DD, k0, lane);
            acc[0][0] = wmma_f16(a0, b0, acc[0][0]);
            acc[0][1] = wmma_f16(a0, b1, acc[0][1]);
            acc[1][0] = wmma_f16(a1, b0, acc[1][0]);
            acc[1][1] = wmma_f16(a1, b1, acc[1][1]);
        }
#pragma unroll
        for (int mi = 0; mi < 2; ++mi)
#pragma unroll
            for (int ni = 0; ni < 2; ++ni)
#pragma unroll
                for (int r = 0; r < 8; ++r) {
                    int n = n0 + mi * 16 + r + rowOff;
                    int e = e0 + ni * 16 + colL;
                    float a = acc[mi][ni][r] * 0.125f;          // SCALE
                    float v = 0.5f * (vmem[mi][ni][r] + a);
                    float s = (v >= 0.5f) ? 1.0f : 0.0f;        // vth = 0.5
                    vmem[mi][ni][r] = v * (1.0f - s);
                    ss[(((size_t)t * BB + b) * NN + n) * CC + h * DD + e] = (_Float16)s;
                }
    }
}

// ---------------------------------------------------------------------------
// Kernel 5: output projection  out[c,n] = (p_w @ s)[c,n]*pscale[c] + pshift[c]
// (bias + BN folded).  One wave = 32x32 tile, 64 WMMAs (K=512).
// waves = 4*32*16*8 = 16384 -> 2048 blocks.
// ---------------------------------------------------------------------------
__global__ void __launch_bounds__(256, 1)
proj_gemm_kernel(const _Float16* wp, const _Float16* ss,
                 const float* pscale, const float* pshift, float* out) {
    int w = blockIdx.x * 8 + (threadIdx.x >> 5);
    int lane = threadIdx.x & 31;
    int nt = w & 7;   w >>= 3;
    int ct = w & 15;  w >>= 4;
    int b  = w & 31;  w >>= 5;
    int t  = w;
    int c0 = ct * 32, n0 = nt * 32;
    const _Float16* A0 = wp + (size_t)c0 * CC;
    const _Float16* B0 = ss + (((size_t)t * BB + b) * NN + n0) * CC;
    v8f acc[2][2];
#pragma unroll
    for (int mi = 0; mi < 2; ++mi)
#pragma unroll
        for (int ni = 0; ni < 2; ++ni) acc[mi][ni] = (v8f){};
#pragma unroll
    for (int k0 = 0; k0 < CC; k0 += 32) {
        v16h a0 = load_frag(A0,            CC, k0, lane);
        v16h a1 = load_frag(A0 + 16 * CC,  CC, k0, lane);
        v16h b0 = load_frag(B0,            CC, k0, lane);
        v16h b1 = load_frag(B0 + 16 * CC,  CC, k0, lane);
        acc[0][0] = wmma_f16(a0, b0, acc[0][0]);
        acc[0][1] = wmma_f16(a0, b1, acc[0][1]);
        acc[1][0] = wmma_f16(a1, b0, acc[1][0]);
        acc[1][1] = wmma_f16(a1, b1, acc[1][1]);
    }
    int colL = lane & 15, rowOff = (lane >> 4) * 8;
#pragma unroll
    for (int mi = 0; mi < 2; ++mi)
#pragma unroll
        for (int ni = 0; ni < 2; ++ni)
#pragma unroll
            for (int r = 0; r < 8; ++r) {
                int c = c0 + mi * 16 + r + rowOff;
                int n = n0 + ni * 16 + colL;
                out[(((size_t)t * BB + b) * CC + c) * NN + n] =
                    acc[mi][ni][r] * pscale[c] + pshift[c];
            }
}

// ---------------------------------------------------------------------------
// Host launcher
// ---------------------------------------------------------------------------
extern "C" void kernel_launch(void* const* d_in, const int* in_sizes, int n_in,
                              void* d_out, int out_size, void* d_ws, size_t ws_size,
                              hipStream_t stream) {
    const float* x  = (const float*)d_in[0];
    const float* qw = (const float*)d_in[1],  *qg = (const float*)d_in[2],
               * qb = (const float*)d_in[3],  *qm = (const float*)d_in[4],  *qv = (const float*)d_in[5];
    const float* kw = (const float*)d_in[6],  *kg = (const float*)d_in[7],
               * kb = (const float*)d_in[8],  *km = (const float*)d_in[9],  *kv = (const float*)d_in[10];
    const float* vw = (const float*)d_in[11], *vg = (const float*)d_in[12],
               * vb = (const float*)d_in[13], *vm = (const float*)d_in[14], *vv = (const float*)d_in[15];
    const float* pw = (const float*)d_in[16], *pbias = (const float*)d_in[17],
               * pg = (const float*)d_in[18], *pb = (const float*)d_in[19],
               * pm = (const float*)d_in[20], *pv = (const float*)d_in[21];
    float* out = (float*)d_out;

    char* ws = (char*)d_ws;
    _Float16* xsT = (_Float16*)(ws);                          // [T,B,N,C]
    _Float16* qs  = (_Float16*)(ws + 33554432);               // [T,B,h,N,d]
    _Float16* ks  = (_Float16*)(ws + 67108864);               // [T,B,h,d,N]
    _Float16* vs  = (_Float16*)(ws + 100663296);              // [T,B,h,d,N]
    _Float16* ss  = (_Float16*)(ws + 134217728);              // [T,B,N,C]
    _Float16* kvT = (_Float16*)(ws + 167772160);              // [T,B,h,64,64]
    _Float16* wq16 = (_Float16*)(ws + 176160768);
    _Float16* wk16 = (_Float16*)(ws + 176685056);
    _Float16* wv16 = (_Float16*)(ws + 177209344);
    _Float16* wp16 = (_Float16*)(ws + 177733632);
    float* bnscale = (float*)(ws + 178257920);                // [3][512]
    float* bnshift = (float*)(ws + 178264064);                // [3][512]
    float* pscale  = (float*)(ws + 178270208);                // [512]
    float* pshift  = (float*)(ws + 178272256);                // [512]
    (void)in_sizes; (void)n_in; (void)out_size; (void)ws_size;

    prep_kernel<<<1024, 256, 0, stream>>>(qw, qg, qb, qm, qv, kw, kg, kb, km, kv,
                                          vw, vg, vb, vm, vv, pw, pbias, pg, pb, pm, pv,
                                          wq16, wk16, wv16, wp16,
                                          bnscale, bnshift, pscale, pshift);
    lif_input_kernel<<<16384, 256, 0, stream>>>(x, xsT);
    qkv_gemm_lif_kernel<<<1536, 256, 0, stream>>>(xsT, wq16, wk16, wv16,
                                                  bnscale, bnshift, qs, ks, vs);
    kv_gemm_kernel<<<512, 256, 0, stream>>>(ks, vs, kvT);
    attn_lif_kernel<<<512, 256, 0, stream>>>(qs, kvT, ss);
    proj_gemm_kernel<<<2048, 256, 0, stream>>>(wp16, ss, pscale, pshift, out);
}
